// BigramLanguageModel_54279796687217
// MI455X (gfx1250) — compile-verified, matrix operand order checked by reading, not run on hardware
//
#include <hip/hip_runtime.h>
#include <hip/hip_bf16.h>

// ---------------- model constants ----------------
#define LNUM 6
#define HNUM 6
#define EDIM 384
#define HSZ  64
#define TSEQ 256
#define BSZ  64
#define VOC  65
#define VPAD 128
#define BT   (BSZ*TSEQ)          // 16384
#define FF   (4*EDIM)            // 1536
#define QKVN (3*EDIM)            // 1152

typedef __bf16 bf16;
typedef __attribute__((ext_vector_type(16))) __bf16 v16bf;
typedef __attribute__((ext_vector_type(8)))  __bf16 v8bf;
typedef __attribute__((ext_vector_type(8)))  float  v8f;
typedef __attribute__((ext_vector_type(4)))  unsigned int v4u;
typedef __attribute__((ext_vector_type(8)))  int v8i;
typedef __attribute__((ext_vector_type(4)))  int v4i;

// TDM availability (this toolchain: 6-arg form, compile-verified)
#if defined(__has_builtin)
#if __has_builtin(__builtin_amdgcn_tensor_load_to_lds) && \
    __has_builtin(__builtin_amdgcn_s_wait_tensorcnt)
#define USE_TDM 1
#endif
#endif
#ifndef USE_TDM
#define USE_TDM 0
#endif

// Build a 16-element bf16 fragment from two contiguous 8-element chunks.
// Per CDNA5 ISA 7.12.2 (16-bit A/B layouts): lane m holds K = 8*half+0..7 in
// regs 0-3 and K = 16+8*half+0..7 in regs 4-7 (half = lane>>4).
__device__ __forceinline__ v16bf make_frag(const bf16* p0, const bf16* p1) {
    v8bf lo = *(const v8bf*)p0;
    v8bf hi = *(const v8bf*)p1;
    v16bf f;
#pragma unroll
    for (int i = 0; i < 8; ++i) { f[i] = lo[i]; f[i + 8] = hi[i]; }
    return f;
}

__device__ __forceinline__ v8f wmma_bf16(v16bf a, v16bf b, v8f c) {
    return __builtin_amdgcn_wmma_f32_16x16x32_bf16(false, a, false, b,
                                                   (short)0, c, false, false);
}

#if USE_TDM
// Issue a TDM 2D tile load: tile_dim0=32 (K elems, contiguous, bf16),
// tile_dim1=128 rows, row stride = Kelems, into LDS at ldsOff (packed).
// D# packing per CDNA5 ISA ch.8 (group0 128b, group1 256b, groups2/3 unused).
__device__ __forceinline__ void tdm_load_tile(const bf16* gptr, unsigned ldsOff,
                                              int Kelems) {
    unsigned long long ga = (unsigned long long)(uintptr_t)gptr;
    v4u g0;
    g0[0] = 1u;                                         // count=1, user mode
    g0[1] = ldsOff;                                     // lds_addr (bytes)
    g0[2] = (unsigned)(ga & 0xFFFFFFFFu);               // global_addr[31:0]
    g0[3] = (unsigned)((ga >> 32) & 0x1FFFFFFu)         // global_addr[56:32]
          | (2u << 30);                                 // type=2 ("image")
    v8i g1;
    g1[0] = (int)(1u << 16);                            // data_size=1 (2B), mask=0
    g1[1] = (int)(((unsigned)Kelems & 0xFFFFu) << 16);  // tensor_dim0[15:0]
    g1[2] = (int)((((unsigned)Kelems >> 16) & 0xFFFFu)  // tensor_dim0[31:16]
          | (((unsigned)BT & 0xFFFFu) << 16));          // tensor_dim1[15:0]
    g1[3] = (int)((((unsigned)BT >> 16) & 0xFFFFu)      // tensor_dim1[31:16]
          | (32u << 16));                               // tile_dim0 = 32
    g1[4] = 128;                                        // tile_dim1=128, tile_dim2=0
    g1[5] = Kelems;                                     // tensor_dim0_stride[31:0]
    g1[6] = 0;                                          // stride hi / dim1_stride lo
    g1[7] = 0;
    v4i gz4;
    gz4[0] = 0; gz4[1] = 0; gz4[2] = 0; gz4[3] = 0;
    v8i gz8;
#pragma unroll
    for (int i = 0; i < 8; ++i) gz8[i] = 0;
    __builtin_amdgcn_tensor_load_to_lds(g0, g1, gz4, gz4, gz8, 0);
}
#endif

// ---------------- weight conversion ----------------
__global__ void cvt_qkv_kernel(const float* __restrict__ Wq,
                               const float* __restrict__ Wk,
                               const float* __restrict__ Wv,
                               bf16* __restrict__ out) {
    int i = blockIdx.x * blockDim.x + threadIdx.x;
    if (i >= LNUM * EDIM * QKVN) return;
    int l = i / (EDIM * QKVN);
    int r = i % (EDIM * QKVN);
    int e = r / QKVN;
    int c = r % QKVN;
    int which = c / EDIM;
    int ec = c % EDIM;
    int h = ec / HSZ, s = ec % HSZ;
    size_t src = (((size_t)l * HNUM + h) * EDIM + e) * HSZ + s;
    float v = (which == 0) ? Wq[src] : (which == 1) ? Wk[src] : Wv[src];
    out[i] = (bf16)v;
}

__global__ void cvt_f32_bf16_kernel(const float* __restrict__ in,
                                    bf16* __restrict__ out, int n) {
    int i = blockIdx.x * blockDim.x + threadIdx.x;
    if (i < n) out[i] = (bf16)in[i];
}

__global__ void cvt_lm_kernel(const float* __restrict__ Wlm, bf16* __restrict__ out) {
    int i = blockIdx.x * blockDim.x + threadIdx.x;
    if (i >= EDIM * VPAD) return;
    int e = i / VPAD, c = i % VPAD;
    out[i] = (bf16)((c < VOC) ? Wlm[e * VOC + c] : 0.f);
}

// ---------------- embedding ----------------
__global__ __launch_bounds__(128)
void embed_kernel(const int* __restrict__ idx, const float* __restrict__ tok,
                  const float* __restrict__ pos, float* __restrict__ x) {
    int row = blockIdx.x;                 // 0..BT-1
    int t = row & (TSEQ - 1);
    int tk = idx[row];
#pragma unroll
    for (int j = 0; j < 3; ++j) {
        int e = threadIdx.x + j * 128;
        x[(size_t)row * EDIM + e] = tok[(size_t)tk * EDIM + e] + pos[(size_t)t * EDIM + e];
    }
}

// ---------------- layernorm (f32 in, bf16 out) ----------------
__global__ __launch_bounds__(128)
void ln_kernel(const float* __restrict__ x, const float* __restrict__ g,
               const float* __restrict__ b, bf16* __restrict__ out) {
    __shared__ float red[128];
    int row = blockIdx.x, tid = threadIdx.x;
    const float* xr = x + (size_t)row * EDIM;
    float v0 = xr[tid], v1 = xr[tid + 128], v2 = xr[tid + 256];
    red[tid] = v0 + v1 + v2;
    __syncthreads();
#pragma unroll
    for (int o = 64; o > 0; o >>= 1) {
        if (tid < o) red[tid] += red[tid + o];
        __syncthreads();
    }
    float mean = red[0] * (1.f / EDIM);
    __syncthreads();
    float d0 = v0 - mean, d1 = v1 - mean, d2 = v2 - mean;
    red[tid] = d0 * d0 + d1 * d1 + d2 * d2;
    __syncthreads();
#pragma unroll
    for (int o = 64; o > 0; o >>= 1) {
        if (tid < o) red[tid] += red[tid + o];
        __syncthreads();
    }
    float rstd = rsqrtf(red[0] * (1.f / EDIM) + 1e-5f);
    bf16* orow = out + (size_t)row * EDIM;
    orow[tid]       = (bf16)(d0 * rstd * g[tid]       + b[tid]);
    orow[tid + 128] = (bf16)(d1 * rstd * g[tid + 128] + b[tid + 128]);
    orow[tid + 256] = (bf16)(d2 * rstd * g[tid + 256] + b[tid + 256]);
}

// ---------------- templated WMMA GEMM ----------------
// C[BT,N] = A[BT,K](bf16,row) x B[K,N](bf16,row)  + mode-specific epilogue.
// 256 threads = 8 waves; block tile 128x128; wave tile 32x64; K step 32;
// ping-pong LDS double buffering, A tile staged by the Tensor Data Mover.
enum { MODE_QKV = 0,     // scatter bf16 into Q/K/V [B][H][T][HS]
       MODE_RESID = 1,   // f32 out = acc + bias + resid   (proj, mlp-down)
       MODE_RELU = 2,    // bf16 out = relu(acc + bias)    (mlp-up)
       MODE_LM = 3 };    // f32 logits (stride VOC), cols < VOC only

template<int N, int K, int MODE>
__global__ __launch_bounds__(256)
void gemm_t(const bf16* __restrict__ A, const bf16* __restrict__ Bm,
            const float* __restrict__ bias, const float* __restrict__ resid,
            float* __restrict__ outF, bf16* __restrict__ outB) {
    __shared__ __attribute__((aligned(16))) bf16 sA[2][128][32];
    __shared__ __attribute__((aligned(16))) bf16 sBt[2][128][32];   // [n][k]

    int tid  = threadIdx.x;
    int wave = tid >> 5;
    int lane = tid & 31;
    int half = (lane >> 4) & 1;
    int mn   = lane & 15;
    int wr   = wave & 3;        // 4 row-wave groups (32 rows each)
    int wc   = wave >> 2;       // 2 col-wave groups (64 cols each)
    int rowBase = blockIdx.y * 128;
    int colBase = blockIdx.x * 128;

    // per-thread staging coordinates
    int aR0  = (tid * 2) >> 2;
    int aK0  = ((tid * 2) & 3) * 8;
    int aR1  = (tid * 2 + 1) >> 2;
    int aK1  = ((tid * 2 + 1) & 3) * 8;
    int bK0  = (tid * 2) >> 4;
    int bN0  = ((tid * 2) & 15) * 8;
    int bK1  = (tid * 2 + 1) >> 4;
    int bN1  = ((tid * 2 + 1) & 15) * 8;

#if USE_TDM
    unsigned sAoff0 = (unsigned)(uintptr_t)
        (__attribute__((address_space(3))) char*)&sA[0][0][0];
    unsigned sAoff1 = (unsigned)(uintptr_t)
        (__attribute__((address_space(3))) char*)&sA[1][0][0];
#endif

    auto stageA = [&](int buf, int k0) {
#if USE_TDM
        if (wave == 0)   // one TDM DMA moves the whole 128x32 tile
            tdm_load_tile(&A[(size_t)rowBase * K + k0],
                          buf ? sAoff1 : sAoff0, K);
#else
        *(v8bf*)&sA[buf][aR0][aK0] =
            *(const v8bf*)&A[(size_t)(rowBase + aR0) * K + k0 + aK0];
        *(v8bf*)&sA[buf][aR1][aK1] =
            *(const v8bf*)&A[(size_t)(rowBase + aR1) * K + k0 + aK1];
#endif
    };
    auto stageB = [&](int buf, int k0) {
        v8bf b0 = *(const v8bf*)&Bm[(size_t)(k0 + bK0) * N + colBase + bN0];
        v8bf b1 = *(const v8bf*)&Bm[(size_t)(k0 + bK1) * N + colBase + bN1];
#pragma unroll
        for (int j = 0; j < 8; ++j) sBt[buf][bN0 + j][bK0] = b0[j];
#pragma unroll
        for (int j = 0; j < 8; ++j) sBt[buf][bN1 + j][bK1] = b1[j];
    };

    v8f acc[2][4];
#pragma unroll
    for (int i = 0; i < 2; ++i)
#pragma unroll
        for (int j = 0; j < 4; ++j) acc[i][j] = (v8f)0.f;

    stageA(0, 0);
    stageB(0, 0);
#if USE_TDM
    if (wave == 0) __builtin_amdgcn_s_wait_tensorcnt(0);
#endif
    __syncthreads();

    for (int k0 = 0; k0 < K; k0 += 32) {
        int cur = (k0 >> 5) & 1;
        if (k0 + 32 < K) {
            stageA(cur ^ 1, k0 + 32);
            stageB(cur ^ 1, k0 + 32);
            if (k0 + 64 < K)   // prefetch B two steps ahead -> global_prefetch_b8
                __builtin_prefetch(&Bm[(size_t)(k0 + 64 + bK0) * N + colBase + bN0], 0, 3);
        }
        v16bf aF[2], bF[4];
#pragma unroll
        for (int rt = 0; rt < 2; ++rt) {
            const bf16* p = &sA[cur][wr * 32 + rt * 16 + mn][0];
            aF[rt] = make_frag(p + 8 * half, p + 16 + 8 * half);
        }
#pragma unroll
        for (int ct = 0; ct < 4; ++ct) {
            const bf16* p = &sBt[cur][wc * 64 + ct * 16 + mn][0];
            bF[ct] = make_frag(p + 8 * half, p + 16 + 8 * half);
        }
#pragma unroll
        for (int rt = 0; rt < 2; ++rt)
#pragma unroll
            for (int ct = 0; ct < 4; ++ct)
                acc[rt][ct] = wmma_bf16(aF[rt], bF[ct], acc[rt][ct]);
#if USE_TDM
        if (k0 + 32 < K && wave == 0) __builtin_amdgcn_s_wait_tensorcnt(0);
#endif
        __syncthreads();
    }

    // ---- specialized epilogue (C layout: VGPR r -> row r+8*half, col = lane&15)
#pragma unroll
    for (int rt = 0; rt < 2; ++rt) {
#pragma unroll
        for (int ct = 0; ct < 4; ++ct) {
            int col  = colBase + wc * 64 + ct * 16 + mn;
            int row0 = rowBase + wr * 32 + rt * 16 + 8 * half;

            if (MODE == MODE_QKV) {
                int which = col / EDIM;
                int ec = col % EDIM;
                int h = ec >> 6, s = ec & 63;
                bf16* dst = outB + (size_t)which * ((size_t)BT * EDIM) + s;
#pragma unroll
                for (int r = 0; r < 8; ++r) {
                    int row = row0 + r;                   // row = b*TSEQ + t
                    int b = row >> 8, t = row & 255;
                    dst[(((size_t)(b * HNUM + h)) * TSEQ + t) * HSZ] =
                        (bf16)acc[rt][ct][r];
                }
            } else if (MODE == MODE_RESID) {
                float bi = bias[col];
#pragma unroll
                for (int r = 0; r < 8; ++r) {
                    size_t o = (size_t)(row0 + r) * N + col;
                    outF[o] = acc[rt][ct][r] + bi + resid[o];
                }
            } else if (MODE == MODE_RELU) {
                float bi = bias[col];
#pragma unroll
                for (int r = 0; r < 8; ++r) {
                    outB[(size_t)(row0 + r) * N + col] =
                        (bf16)fmaxf(acc[rt][ct][r] + bi, 0.f);
                }
            } else { // MODE_LM
                if (col < VOC) {
                    float bi = bias[col];
#pragma unroll
                    for (int r = 0; r < 8; ++r)
                        outF[(size_t)(row0 + r) * VOC + col] = acc[rt][ct][r] + bi;
                }
            }
        }
    }
}

// ---------------- attention ----------------
// One wave per (b, h, 16-row query block). Q/K/V in [B][H][T][HS] bf16.
__global__ __launch_bounds__(32)
void attn_kernel(const bf16* __restrict__ Q, const bf16* __restrict__ Kb,
                 const bf16* __restrict__ Vb, bf16* __restrict__ attn_out) {
    __shared__ __attribute__((aligned(16))) float sS[16 * TSEQ];   // scores
    __shared__ __attribute__((aligned(16))) bf16  sP[16 * TSEQ];   // probs
    __shared__ __attribute__((aligned(16))) bf16  sVt[HSZ][32];    // V^T chunk

    int qb = blockIdx.x & 15;
    int h  = (blockIdx.x >> 4) % HNUM;
    int b  = blockIdx.x / (16 * HNUM);
    int lane = threadIdx.x & 31;
    int half = (lane >> 4) & 1;
    int mn   = lane & 15;
    int q0   = qb * 16;
    const float scale = 0.05103103630798287f;   // 1/sqrt(E)  (reference uses E, not HS)
    size_t base = ((size_t)(b * HNUM + h)) * TSEQ * HSZ;

    v16bf qF[2];
#pragma unroll
    for (int ks = 0; ks < 2; ++ks) {
        const bf16* p = Q + base + (size_t)(q0 + mn) * HSZ + ks * 32;
        qF[ks] = make_frag(p + 8 * half, p + 16 + 8 * half);
    }

    // scores = Q K^T * scale over causal key blocks
    for (int j = 0; j <= qb; ++j) {
        v8f sc = (v8f)0.f;
#pragma unroll
        for (int ks = 0; ks < 2; ++ks) {
            const bf16* p = Kb + base + (size_t)(j * 16 + mn) * HSZ + ks * 32;
            v16bf kF = make_frag(p + 8 * half, p + 16 + 8 * half);
            sc = wmma_bf16(qF[ks], kF, sc);
        }
#pragma unroll
        for (int r = 0; r < 8; ++r)
            sS[(r + 8 * half) * TSEQ + j * 16 + mn] = sc[r] * scale;
    }
    __syncthreads();

    int nkpad = ((qb + 1) * 16 + 31) & ~31;
    if (nkpad > TSEQ) nkpad = TSEQ;

    // causal softmax: thread m handles row m
    if (lane < 16) {
        int m = lane;
        int nv = q0 + m + 1;
        float* row = &sS[m * TSEQ];
        float mx = -1e30f;
        for (int c = 0; c < nv; ++c) mx = fmaxf(mx, row[c]);
        float s = 0.f;
        for (int c = 0; c < nv; ++c) s += __expf(row[c] - mx);
        float inv = 1.f / s;
        bf16* prow = &sP[m * TSEQ];
        for (int c = 0; c < nv; ++c) prow[c] = (bf16)(__expf(row[c] - mx) * inv);
        for (int c = nv; c < nkpad; ++c) prow[c] = (bf16)0.f;
    }
    __syncthreads();

    // O = P V
    v8f o[4];
#pragma unroll
    for (int ct = 0; ct < 4; ++ct) o[ct] = (v8f)0.f;

    for (int kb = 0; kb < nkpad; kb += 32) {
        {   // stage 32 key rows of V transposed: sVt[s][key_local]
            int key = kb + lane;
            const bf16* vrow = Vb + base + (size_t)key * HSZ;
#pragma unroll
            for (int s8 = 0; s8 < 8; ++s8) {
                v8bf v = *(const v8bf*)&vrow[s8 * 8];
#pragma unroll
                for (int j = 0; j < 8; ++j) sVt[s8 * 8 + j][lane] = v[j];
            }
        }
        __syncthreads();
        const bf16* pp = &sP[mn * TSEQ + kb];
        v16bf pF = make_frag(pp + 8 * half, pp + 16 + 8 * half);
#pragma unroll
        for (int ct = 0; ct < 4; ++ct) {
            const bf16* vp = &sVt[ct * 16 + mn][0];
            v16bf vF = make_frag(vp + 8 * half, vp + 16 + 8 * half);
            o[ct] = wmma_bf16(pF, vF, o[ct]);
        }
        __syncthreads();
    }

    // write attn_out[b][t][h*HS + s]  (heads concatenated)
#pragma unroll
    for (int ct = 0; ct < 4; ++ct) {
#pragma unroll
        for (int r = 0; r < 8; ++r) {
            int t = q0 + r + 8 * half;
            int col = h * HSZ + ct * 16 + mn;
            attn_out[((size_t)(b * TSEQ + t)) * EDIM + col] = (bf16)o[ct][r];
        }
    }
}

// ---------------- loss ----------------
__global__ __launch_bounds__(256)
void loss_rows_kernel(const float* __restrict__ logits, const int* __restrict__ targets,
                      float* __restrict__ rowloss) {
    int row = blockIdx.x * blockDim.x + threadIdx.x;
    if (row >= BT) return;
    const float* lr = logits + (size_t)row * VOC;
    float mx = -1e30f;
    for (int c = 0; c < VOC; ++c) mx = fmaxf(mx, lr[c]);
    float s = 0.f;
    for (int c = 0; c < VOC; ++c) s += __expf(lr[c] - mx);
    float lse = __logf(s) + mx;
    rowloss[row] = lse - lr[targets[row]];
}

__global__ __launch_bounds__(256)
void loss_reduce_kernel(const float* __restrict__ rowloss, float* __restrict__ out) {
    __shared__ float red[256];
    int tid = threadIdx.x;
    float s = 0.f;
    for (int i = tid; i < BT; i += 256) s += rowloss[i];
    red[tid] = s;
    __syncthreads();
#pragma unroll
    for (int o = 128; o > 0; o >>= 1) {
        if (tid < o) red[tid] += red[tid + o];
        __syncthreads();
    }
    if (tid == 0) out[0] = red[0] * (1.f / BT);
}

// ---------------- host launch ----------------
extern "C" void kernel_launch(void* const* d_in, const int* in_sizes, int n_in,
                              void* d_out, int out_size, void* d_ws, size_t ws_size,
                              hipStream_t stream) {
    (void)in_sizes; (void)n_in; (void)out_size; (void)ws_size;

    const int*   idx     = (const int*)  d_in[0];
    const int*   targets = (const int*)  d_in[1];
    const float* tok_emb = (const float*)d_in[2];
    const float* pos_emb = (const float*)d_in[3];
    const float* Wq      = (const float*)d_in[4];
    const float* Wk      = (const float*)d_in[5];
    const float* Wv      = (const float*)d_in[6];
    const float* Wproj   = (const float*)d_in[7];
    const float* bproj   = (const float*)d_in[8];
    const float* ln1_g   = (const float*)d_in[9];
    const float* ln1_b   = (const float*)d_in[10];
    const float* ln2_g   = (const float*)d_in[11];
    const float* ln2_b   = (const float*)d_in[12];
    const float* W1      = (const float*)d_in[13];
    const float* b1      = (const float*)d_in[14];
    const float* W2      = (const float*)d_in[15];
    const float* b2      = (const float*)d_in[16];
    const float* lnf_g   = (const float*)d_in[17];
    const float* lnf_b   = (const float*)d_in[18];
    const float* Wlm     = (const float*)d_in[19];
    const float* blm     = (const float*)d_in[20];

    float* logits = (float*)d_out;                 // [BT, V]
    float* lossp  = (float*)d_out + (size_t)BT * VOC;

    size_t off = 0;
    auto carve = [&](size_t bytes) -> char* {
        char* p = (char*)d_ws + off;
        off += (bytes + 255) & ~(size_t)255;
        return p;
    };
    bf16*  Wqkvb  = (bf16*) carve((size_t)LNUM * EDIM * QKVN * 2);
    bf16*  Wprojb = (bf16*) carve((size_t)LNUM * EDIM * EDIM * 2);
    bf16*  W1b    = (bf16*) carve((size_t)LNUM * EDIM * FF   * 2);
    bf16*  W2b    = (bf16*) carve((size_t)LNUM * FF   * EDIM * 2);
    bf16*  WlmP   = (bf16*) carve((size_t)EDIM * VPAD * 2);
    float* x      = (float*)carve((size_t)BT * EDIM * 4);
    bf16*  hbuf   = (bf16*) carve((size_t)BT * EDIM * 2);
    bf16*  qkv    = (bf16*) carve((size_t)3 * BT * EDIM * 2);
    bf16*  attnb  = (bf16*) carve((size_t)BT * EDIM * 2);
    bf16*  mlph   = (bf16*) carve((size_t)BT * FF   * 2);
    float* rowls  = (float*)carve((size_t)BT * 4);

    {
        int n = LNUM * EDIM * QKVN;
        cvt_qkv_kernel<<<(n + 255) / 256, 256, 0, stream>>>(Wq, Wk, Wv, Wqkvb);
        n = LNUM * EDIM * EDIM;
        cvt_f32_bf16_kernel<<<(n + 255) / 256, 256, 0, stream>>>(Wproj, Wprojb, n);
        n = LNUM * EDIM * FF;
        cvt_f32_bf16_kernel<<<(n + 255) / 256, 256, 0, stream>>>(W1, W1b, n);
        cvt_f32_bf16_kernel<<<(n + 255) / 256, 256, 0, stream>>>(W2, W2b, n);
        n = EDIM * VPAD;
        cvt_lm_kernel<<<(n + 255) / 256, 256, 0, stream>>>(Wlm, WlmP);
    }

    embed_kernel<<<BT, 128, 0, stream>>>(idx, tok_emb, pos_emb, x);

    bf16* Qb   = qkv;
    bf16* Kbuf = qkv + (size_t)BT * EDIM;
    bf16* Vbuf = qkv + (size_t)2 * BT * EDIM;

    for (int l = 0; l < LNUM; ++l) {
        ln_kernel<<<BT, 128, 0, stream>>>(x, ln1_g + l * EDIM, ln1_b + l * EDIM, hbuf);
        gemm_t<QKVN, EDIM, MODE_QKV><<<dim3(QKVN / 128, BT / 128), 256, 0, stream>>>(
            hbuf, Wqkvb + (size_t)l * EDIM * QKVN, nullptr, nullptr, nullptr, qkv);
        attn_kernel<<<BSZ * HNUM * (TSEQ / 16), 32, 0, stream>>>(Qb, Kbuf, Vbuf, attnb);
        gemm_t<EDIM, EDIM, MODE_RESID><<<dim3(EDIM / 128, BT / 128), 256, 0, stream>>>(
            attnb, Wprojb + (size_t)l * EDIM * EDIM, bproj + l * EDIM, x, x, nullptr);
        ln_kernel<<<BT, 128, 0, stream>>>(x, ln2_g + l * EDIM, ln2_b + l * EDIM, hbuf);
        gemm_t<FF, EDIM, MODE_RELU><<<dim3(FF / 128, BT / 128), 256, 0, stream>>>(
            hbuf, W1b + (size_t)l * EDIM * FF, b1 + l * FF, nullptr, nullptr, mlph);
        gemm_t<EDIM, FF, MODE_RESID><<<dim3(EDIM / 128, BT / 128), 256, 0, stream>>>(
            mlph, W2b + (size_t)l * FF * EDIM, b2 + l * EDIM, x, x, nullptr);
    }

    ln_kernel<<<BT, 128, 0, stream>>>(x, lnf_g, lnf_b, hbuf);
    gemm_t<VPAD, EDIM, MODE_LM><<<dim3(VPAD / 128, BT / 128), 256, 0, stream>>>(
        hbuf, WlmP, blm, nullptr, logits, nullptr);

    loss_rows_kernel<<<BT / 256, 256, 0, stream>>>(logits, targets, rowls);
    loss_reduce_kernel<<<1, 256, 0, stream>>>(rowls, lossp);
}
